// Network_43817256353809
// MI455X (gfx1250) — matrix operand order repaired and checked by reading, not compiled, not run on hardware
//
#include <hip/hip_runtime.h>
#include <hip/hip_bf16.h>
#include <math.h>
#include <stdint.h>

typedef __attribute__((ext_vector_type(16))) _Float16 v16h;
typedef __attribute__((ext_vector_type(8)))  _Float16 v8h;
typedef __attribute__((ext_vector_type(8)))  float    v8f;
typedef __attribute__((ext_vector_type(4)))  unsigned int u32x4;
typedef __attribute__((ext_vector_type(4)))  int i32x4;
typedef __attribute__((ext_vector_type(8)))  int i32x8;

union Frag { v16h v; v8h h[2]; _Float16 e[16]; };

#define WMMA16(a,b,c) __builtin_amdgcn_wmma_f32_16x16x32_f16(false,(a),false,(b),(short)0,(c),false,false)

#ifndef __has_builtin
#define __has_builtin(x) 0
#endif
#if __has_builtin(__builtin_amdgcn_tensor_load_to_lds) && __has_builtin(__builtin_amdgcn_s_wait_tensorcnt)
#define HAVE_TDM 1
#else
#define HAVE_TDM 0
#endif

// ---- problem dims ----
#define BB   128
#define SS   16
#define NN   4
#define CIN  6
#define CNN  128
#define HID  256
#define LAT  784
#define LATP 800          // LAT padded to multiple of 32
#define NS   (BB*NN*SS)   // 8192 conv samples
#define NCOLS (NS*49)     // 401408 GEMM columns for conv
#define ROWS (BB*NN)      // 512 recurrent rows

#if HAVE_TDM
// --------------------------------------------------------------------------
// Tensor Data Mover: async 2D tile load (f16 elements) global -> LDS.
// D# per CDNA5 ISA (2D, no pad/iterate).
__device__ __forceinline__ void tdm_load2d(unsigned lds_byte_off, const _Float16* gptr,
                                           unsigned tile_w, unsigned tile_h,
                                           unsigned stride_elems,
                                           unsigned dim0, unsigned dim1) {
  unsigned long long ga = (unsigned long long)(uintptr_t)gptr;
  u32x4 g0;
  g0[0] = 1u;                                             // count=1, user descriptor
  g0[1] = lds_byte_off;                                   // lds_addr (bytes)
  g0[2] = (unsigned)(ga & 0xffffffffu);                   // global_addr[31:0]
  g0[3] = (unsigned)((ga >> 32) & 0x01ffffffu) | (2u << 30); // addr[56:32] | type=2
  i32x8 g1;
  g1[0] = (int)(1u << 16);                                // data_size=1 -> 2 bytes
  g1[1] = (int)((dim0 & 0xffffu) << 16);                  // tensor_dim0[15:0] @bit48
  g1[2] = (int)((dim0 >> 16) | ((dim1 & 0xffffu) << 16)); // dim0[31:16] | dim1[15:0]
  g1[3] = (int)(((dim1 >> 16) & 0xffffu) | (tile_w << 16)); // dim1[31:16] | tile_dim0
  g1[4] = (int)(tile_h & 0xffffu);                        // tile_dim1 (tile_dim2=0)
  g1[5] = (int)stride_elems;                              // tensor_dim0_stride[31:0]
  g1[6] = 0;                                              // stride0[47:32] | stride1 lo
  g1[7] = 0;
  i32x4 z4 = {0, 0, 0, 0};
#if __clang_major__ >= 23
  i32x8 z8 = {0, 0, 0, 0, 0, 0, 0, 0};
  __builtin_amdgcn_tensor_load_to_lds(g0, g1, z4, z4, z8, 0);
#else
  __builtin_amdgcn_tensor_load_to_lds(g0, g1, z4, z4, 0);
#endif
}
#endif // HAVE_TDM

// --------------------------------------------------------------------------
// f32 -> f16 weight conversion with zero K-padding (also used as a zero-fill)
__global__ void k_cvt(const float* __restrict__ src, _Float16* __restrict__ dst,
                      int rows, int K, int ldk) {
  int i = blockIdx.x * 256 + threadIdx.x;
  if (i >= rows * ldk) return;
  int r = i / ldk, k = i % ldk;
  dst[i] = (k < K) ? (_Float16)src[r * K + k] : (_Float16)0.0f;
}

// --------------------------------------------------------------------------
// conv0: 6ch 9x9 -> 128ch 7x7 valid.  GEMM M=128, K=54(pad 64), N=401408.
__global__ void k_conv0(const float* __restrict__ obs,
                        const _Float16* __restrict__ W,   // [128][64]
                        const float* __restrict__ bias,
                        _Float16* __restrict__ Y) {       // [NS][128][49]
  __shared__ _Float16 ldsB[64 * 32];
  int tid = threadIdx.x, wave = tid >> 5, lane = tid & 31;
  int n0 = blockIdx.x * 64, m0 = wave * 16;
  int hl = (lane < 16) ? 0 : 1, l15 = lane & 15;
  v8f acc[4] = {};
  for (int k0 = 0; k0 < 64; k0 += 32) {
    for (int i = tid; i < 64 * 32; i += 256) {
      int col = i >> 5, kk = i & 31, k = k0 + kk;
      _Float16 val = (_Float16)0.0f;
      if (k < 54) {
        int ci = k / 9, rr = (k % 9) / 3, cc = k % 3;
        int n = n0 + col, j = n / 49, pos = n % 49;
        int b = j >> 6, nag = (j >> 4) & 3, s = j & 15;
        int iy = pos / 7 + rr, ix = pos % 7 + cc;
        val = (_Float16)obs[(((b * SS + s) * NN + nag) * CIN + ci) * 81 + iy * 9 + ix];
      }
      ldsB[col * 32 + kk] = val;
    }
    __syncthreads();
    Frag a;
    const _Float16* ap = W + (m0 + l15) * 64 + hl * 8 + k0;
    a.h[0] = *(const v8h*)(ap); a.h[1] = *(const v8h*)(ap + 16);
#pragma unroll
    for (int t = 0; t < 4; ++t) {
      Frag bf;
      const _Float16* bp = &ldsB[(t * 16 + l15) * 32 + hl * 8];
      bf.h[0] = *(const v8h*)(bp); bf.h[1] = *(const v8h*)(bp + 16);
      acc[t] = WMMA16(a.v, bf.v, acc[t]);
    }
    __syncthreads();
  }
  int mbase = m0 + hl * 8;
#pragma unroll
  for (int t = 0; t < 4; ++t) {
    int n = n0 + t * 16 + l15, sample = n / 49, pos = n % 49;
#pragma unroll
    for (int j = 0; j < 8; ++j) {
      int ch = mbase + j;
      float v = acc[t][j] + bias[ch];
      v = v > 0.f ? v : 0.f;
      Y[(sample * CNN + ch) * 49 + pos] = (_Float16)v;
    }
  }
}

// --------------------------------------------------------------------------
// residual 3x3 conv, pad=1, 128->128.  GEMM M=128, K=1152, N=401408.
// A (weight) slab TDM-loaded into LDS by wave 0; B staged via im2col gather.
__global__ void k_conv3x3(const _Float16* __restrict__ W,    // [128][1152]
                          const float* __restrict__ bias,
                          const _Float16* __restrict__ X,    // [NS][128][49]
                          const _Float16* __restrict__ RES,  // optional residual
                          _Float16* __restrict__ Y) {
  __shared__ _Float16 ldsB[64 * 32];
#if HAVE_TDM
  __shared__ _Float16 ldsA[128 * 32];
#endif
  int tid = threadIdx.x, wave = tid >> 5, lane = tid & 31;
  int n0 = blockIdx.x * 64, m0 = wave * 16;
  int hl = (lane < 16) ? 0 : 1, l15 = lane & 15;
  v8f acc[4] = {};

  // K-invariant im2col geometry for the 8 staged elements of this thread
  int skk[8], ssamp[8], soy[8], sox[8];
#pragma unroll
  for (int j = 0; j < 8; ++j) {
    int i = tid + j * 256;
    skk[j] = i & 31;
    int n = n0 + (i >> 5);
    int sample = n / 49, pos = n - sample * 49;
    ssamp[j] = sample;
    soy[j] = pos / 7;
    sox[j] = pos - soy[j] * 7;
  }
#if HAVE_TDM
  unsigned aoff = (unsigned)(uintptr_t)&ldsA[0];
#else
  const _Float16* ap0 = W + (m0 + l15) * 1152 + hl * 8;
#endif

  for (int k0 = 0; k0 < 1152; k0 += 32) {
#if HAVE_TDM
    if (wave == 0)  // async: pull 128x32 f16 weight slab into LDS via TDM
      tdm_load2d(aoff, W + k0, 32, 128, 1152, 1152, 128);
#endif
#pragma unroll
    for (int j = 0; j < 8; ++j) {
      int k = k0 + skk[j];
      int ci = k / 9, rem = k - ci * 9;
      int rr = rem / 3, cc = rem - rr * 3;
      int iy = soy[j] + rr - 1, ix = sox[j] + cc - 1;
      _Float16 val = (_Float16)0.0f;
      if (iy >= 0 && iy < 7 && ix >= 0 && ix < 7)
        val = X[(ssamp[j] * CNN + ci) * 49 + iy * 7 + ix];
      ldsB[tid + j * 256] = val;
    }
#if HAVE_TDM
    if (wave == 0) __builtin_amdgcn_s_wait_tensorcnt((short)0);
#endif
    __syncthreads();
    Frag a;
#if HAVE_TDM
    const _Float16* apl = &ldsA[(m0 + l15) * 32 + hl * 8];
    a.h[0] = *(const v8h*)(apl); a.h[1] = *(const v8h*)(apl + 16);
#else
    a.h[0] = *(const v8h*)(ap0 + k0);
    a.h[1] = *(const v8h*)(ap0 + k0 + 16);
#endif
#pragma unroll
    for (int t = 0; t < 4; ++t) {
      Frag bf;
      const _Float16* bp = &ldsB[(t * 16 + l15) * 32 + hl * 8];
      bf.h[0] = *(const v8h*)(bp); bf.h[1] = *(const v8h*)(bp + 16);
      acc[t] = WMMA16(a.v, bf.v, acc[t]);
    }
    __syncthreads();
  }
  int mbase = m0 + hl * 8;
#pragma unroll
  for (int t = 0; t < 4; ++t) {
    int n = n0 + t * 16 + l15, sample = n / 49, pos = n % 49;
#pragma unroll
    for (int j = 0; j < 8; ++j) {
      int ch = mbase + j;
      float v = acc[t][j] + bias[ch];
      if (RES) v += (float)RES[(sample * CNN + ch) * 49 + pos];
      v = v > 0.f ? v : 0.f;
      Y[(sample * CNN + ch) * 49 + pos] = (_Float16)v;
    }
  }
}

// --------------------------------------------------------------------------
// 1x1 conv 128->16, writes latent f16 [NS][800] (k = ch*49+pos).
__global__ void k_conv1x1(const _Float16* __restrict__ W,   // [16][128]
                          const float* __restrict__ bias,
                          const _Float16* __restrict__ X,   // [NS][128][49]
                          _Float16* __restrict__ L) {       // [NS][800]
  int wave = threadIdx.x >> 5, lane = threadIdx.x & 31;
  int tile = blockIdx.x * 8 + wave;
  int n0 = tile * 64;
  int hl = (lane < 16) ? 0 : 1, l15 = lane & 15;
  v8f acc[4] = {};
  for (int k0 = 0; k0 < 128; k0 += 32) {
    Frag a;
    const _Float16* ap = W + l15 * 128 + hl * 16 + k0;
    a.h[0] = *(const v8h*)(ap); a.h[1] = *(const v8h*)(ap + 8);
#pragma unroll
    for (int t = 0; t < 4; ++t) {
      int n = n0 + t * 16 + l15, sample = n / 49, pos = n % 49;
      int kb = k0 + hl * 16;
      Frag bf;
#pragma unroll
      for (int q = 0; q < 16; ++q)
        bf.e[q] = X[(sample * CNN + kb + q) * 49 + pos];
      acc[t] = WMMA16(a.v, bf.v, acc[t]);
    }
  }
  int mbase = hl * 8;
#pragma unroll
  for (int t = 0; t < 4; ++t) {
    int n = n0 + t * 16 + l15, sample = n / 49, pos = n % 49;
#pragma unroll
    for (int j = 0; j < 8; ++j) {
      int ch = mbase + j;
      float v = acc[t][j] + bias[ch];
      v = v > 0.f ? v : 0.f;
      L[sample * LATP + ch * 49 + pos] = (_Float16)v;
    }
  }
}

// --------------------------------------------------------------------------
// Generic WMMA GEMM: C[n][m] = sum_k A[m][k]*B[n][k] (+bias[m]); optional f16 mirror.
// A row-major [M][lda]; B column-contiguous [N][ldb] (K contiguous per column).
// A slabs are TDM double-buffered per wave (async pipeline on TENSORcnt).
__global__ void k_gemm(const _Float16* __restrict__ A, int lda,
                       const _Float16* __restrict__ B, int ldb,
                       const float* __restrict__ bias,
                       float* __restrict__ C, int ldc,
                       _Float16* __restrict__ C16, int ldc16,
                       int tiles_m, int tiles_n, int K, int relu) {
#if HAVE_TDM
  __shared__ _Float16 ldsA[8 * 2 * 512];   // per wave: 2 x (16 rows x 32 halfs)
#endif
  int wave = threadIdx.x >> 5, lane = threadIdx.x & 31;
  int tile = blockIdx.x * 8 + wave;
  if (tile >= tiles_m * tiles_n) return;
  int m0 = (tile % tiles_m) * 16;
  int n0 = (tile / tiles_m) * 64;
  int hl = (lane < 16) ? 0 : 1, l15 = lane & 15;
  v8f acc[4] = {};
  const _Float16* bp[4];
#pragma unroll
  for (int t = 0; t < 4; ++t) bp[t] = B + (n0 + t * 16 + l15) * ldb + hl * 8;

#if HAVE_TDM
  const _Float16* agbl = A + m0 * lda;            // 16-row slab base
  unsigned abase = (unsigned)(uintptr_t)&ldsA[wave * 1024];  // bytes (2KB/wave)
  tdm_load2d(abase, agbl, 32, 16, lda, lda, 16);  // prime buffer 0
  int ib = 0;
#else
  const _Float16* ap = A + (m0 + l15) * lda + hl * 8;
#endif

  for (int k0 = 0; k0 < K; k0 += 32) {
    Frag a;
#if HAVE_TDM
    if (k0 + 32 < K) {                            // issue next slab, then wait
      tdm_load2d(abase + (unsigned)((ib ^ 1) * 1024), agbl + k0 + 32, 32, 16, lda, lda, 16);
      __builtin_amdgcn_s_wait_tensorcnt((short)1);   // current done, next in flight
    } else {
      __builtin_amdgcn_s_wait_tensorcnt((short)0);
    }
    const _Float16* apl = &ldsA[wave * 1024 + ib * 512 + l15 * 32 + hl * 8];
    a.h[0] = *(const v8h*)(apl); a.h[1] = *(const v8h*)(apl + 16);
    ib ^= 1;
#else
    a.h[0] = *(const v8h*)(ap + k0);
    a.h[1] = *(const v8h*)(ap + k0 + 16);
#endif
    __builtin_prefetch(bp[0] + k0 + 32, 0, 1);
#pragma unroll
    for (int t = 0; t < 4; ++t) {
      Frag bf;
      bf.h[0] = *(const v8h*)(bp[t] + k0);
      bf.h[1] = *(const v8h*)(bp[t] + k0 + 16);
      acc[t] = WMMA16(a.v, bf.v, acc[t]);
    }
  }
  int mbase = m0 + hl * 8;
#pragma unroll
  for (int t = 0; t < 4; ++t) {
    int n = n0 + t * 16 + l15;
    float* cp = C + n * ldc + mbase;
#pragma unroll
    for (int j = 0; j < 8; ++j) {
      float v = acc[t][j];
      if (bias) v += bias[mbase + j];
      if (relu) v = v > 0.f ? v : 0.f;
      cp[j] = v;
      if (C16) C16[n * ldc16 + mbase + j] = (_Float16)v;
    }
  }
}

// --------------------------------------------------------------------------
// GRU gate elementwise.  cmS points at comm_mask for this step (stride 256/b).
__global__ void k_gru(const float* __restrict__ gi, const float* __restrict__ gh,
                      float* __restrict__ h, _Float16* __restrict__ h16,
                      const unsigned char* __restrict__ cmS, int use_mask) {
  int i = blockIdx.x * 256 + threadIdx.x;
  if (i >= ROWS * HID) return;
  int row = i / HID, c = i % HID;
  float hv = h[i];
  float ir = gi[row * 768 + c], iz = gi[row * 768 + 256 + c], ig = gi[row * 768 + 512 + c];
  float hr = gh[row * 768 + c], hz = gh[row * 768 + 256 + c], hg = gh[row * 768 + 512 + c];
  float r = 1.f / (1.f + __expf(-(ir + hr)));
  float z = 1.f / (1.f + __expf(-(iz + hz)));
  float g = tanhf(ig + r * hg);
  float hn = (1.f - z) * g + z * hv;
  if (use_mask) {
    int b = row >> 2, n = row & 3;
    const unsigned char* p = cmS + b * 256 + n * 4;
    int cnt = p[0] + p[1] + p[2] + p[3];
    if (cnt <= 1) hn = hv;   // only agents with >1 neighbor take the update
  }
  h[i] = hn;
  h16[i] = (_Float16)hn;
}

// --------------------------------------------------------------------------
// Masked 2-head attention over N=4 agents (tiny -> scalar VALU).
__global__ void k_attn(const float* __restrict__ q, const float* __restrict__ kk,
                       const float* __restrict__ v, const unsigned char* __restrict__ cmS,
                       float* __restrict__ ctx, _Float16* __restrict__ ctx16) {
  __shared__ float qs[4][64], ks2[4][64], vs[4][64], sc[4][4], aw[4][4];
  int b = blockIdx.x >> 1, hd = blockIdx.x & 1;
  int tid = threadIdx.x, n = tid >> 6, d = tid & 63;
  int row = b * 4 + n, f = hd * 64 + d;
  qs[n][d]  = q[row * 128 + f];
  ks2[n][d] = kk[row * 128 + f];
  vs[n][d]  = v[row * 128 + f];
  __syncthreads();
  if (tid < 16) {
    int n2 = tid >> 2, m = tid & 3;
    float s = 0.f;
    for (int e = 0; e < 64; ++e) s += qs[n2][e] * ks2[m][e];
    s *= 0.125f;
    if (!cmS[b * 256 + n2 * 4 + m]) s = -1e9f;   // amask = ~comm_mask
    sc[n2][m] = s;
  }
  __syncthreads();
  if (tid < 4) {
    float mx = fmaxf(fmaxf(sc[tid][0], sc[tid][1]), fmaxf(sc[tid][2], sc[tid][3]));
    float sum = 0.f;
    for (int m = 0; m < 4; ++m) { float e = __expf(sc[tid][m] - mx); aw[tid][m] = e; sum += e; }
    for (int m = 0; m < 4; ++m) aw[tid][m] /= sum;
  }
  __syncthreads();
  float o = 0.f;
  for (int m = 0; m < 4; ++m) o += aw[n][m] * vs[m][d];
  ctx[row * 128 + f] = o;
  ctx16[row * 128 + f] = (_Float16)o;
}

// --------------------------------------------------------------------------
__global__ void k_inith(const float* __restrict__ hidden, float* __restrict__ h,
                        _Float16* __restrict__ h16) {
  int i = blockIdx.x * 256 + threadIdx.x;
  if (i >= ROWS * HID) return;
  float v = hidden[i]; h[i] = v; h16[i] = (_Float16)v;
}

__global__ void k_hbuf(const float* __restrict__ h, float* __restrict__ hbuf, int s) {
  int i = blockIdx.x * 256 + threadIdx.x;
  if (i >= BB * HID) return;
  int b = i / HID, c = i % HID;
  hbuf[(s * BB + b) * HID + c] = h[(b * NN) * HID + c];   // agent-0 rows
}

// --------------------------------------------------------------------------
// Final head: per-b MLP(agent0, selected step) + dueling combine.
__global__ void k_head(const _Float16* __restrict__ L, const float* __restrict__ hbuf,
                       const int* __restrict__ steps,
                       const float* __restrict__ m0w, const float* __restrict__ m0b,
                       const float* __restrict__ m1w, const float* __restrict__ m1b,
                       const float* __restrict__ advw, const float* __restrict__ advb,
                       const float* __restrict__ stw, const float* __restrict__ stb,
                       float* __restrict__ out) {
  __shared__ float t[HID];
  __shared__ float msh, advs[5], svsh;
  int b = blockIdx.x, tid = threadIdx.x;
  int idx = steps[b] - 1;
  const _Float16* lat = L + (b * 64 + idx) * LATP;   // sample (b, agent0, step idx)
  float acc = m0b[tid];
  for (int k = 0; k < LAT; ++k) acc += (float)lat[k] * m0w[tid * LAT + k];
  t[tid] = acc;
  __syncthreads();
  if (tid == 0) {
    float s = m1b[0];
    for (int i = 0; i < HID; ++i) s += t[i] * m1w[i];
    msh = 1.f / (1.f + __expf(-s));
  }
  __syncthreads();
  const float* hs = hbuf + (idx * BB + b) * HID;
  if (tid < 5) {
    float a = advb[tid];
    for (int c = 0; c < HID; ++c) a += hs[c] * advw[tid * 257 + c];
    a += msh * advw[tid * 257 + 256];
    advs[tid] = a;
  }
  if (tid == 8) {
    float s = stb[0];
    for (int c = 0; c < HID; ++c) s += hs[c] * stw[c];
    svsh = s;
  }
  __syncthreads();
  if (tid == 0) {
    float mean = (advs[0] + advs[1] + advs[2] + advs[3] + advs[4]) * 0.2f;
    for (int a = 0; a < 5; ++a) out[b * 5 + a] = svsh + advs[a] - mean;
  }
}

// ==========================================================================
extern "C" void kernel_launch(void* const* d_in, const int* in_sizes, int n_in,
                              void* d_out, int out_size, void* d_ws, size_t ws_size,
                              hipStream_t stream) {
  (void)in_sizes; (void)n_in; (void)out_size; (void)ws_size;
  const float* obs    = (const float*)d_in[0];
  const int*   steps  = (const int*)d_in[1];
  const float* hidden = (const float*)d_in[2];
  const unsigned char* cm = (const unsigned char*)d_in[3];
  const float* conv0_w = (const float*)d_in[4];
  const float* conv0_b = (const float*)d_in[5];
  const float* res_w[6]; const float* res_b[6];
  for (int i = 0; i < 6; ++i) { res_w[i] = (const float*)d_in[6 + 2 * i]; res_b[i] = (const float*)d_in[7 + 2 * i]; }
  const float* conv1_w = (const float*)d_in[18];
  const float* conv1_b = (const float*)d_in[19];
  const float* mlp0_w = (const float*)d_in[20];
  const float* mlp0_b = (const float*)d_in[21];
  const float* mlp1_w = (const float*)d_in[22];
  const float* mlp1_b = (const float*)d_in[23];
  const float* rec_wih = (const float*)d_in[24];
  const float* rec_whh = (const float*)d_in[25];
  const float* rec_bih = (const float*)d_in[26];
  const float* rec_bhh = (const float*)d_in[27];
  const float* wq_w = (const float*)d_in[28];
  const float* wq_b = (const float*)d_in[29];
  const float* wk_w = (const float*)d_in[30];
  const float* wk_b = (const float*)d_in[31];
  const float* wv_w = (const float*)d_in[32];
  const float* wv_b = (const float*)d_in[33];
  const float* wo_w = (const float*)d_in[34];
  const float* upd_wih = (const float*)d_in[35];
  const float* upd_whh = (const float*)d_in[36];
  const float* upd_bih = (const float*)d_in[37];
  const float* upd_bhh = (const float*)d_in[38];
  const float* adv_w = (const float*)d_in[39];
  const float* adv_b = (const float*)d_in[40];
  const float* state_w = (const float*)d_in[41];
  const float* state_b = (const float*)d_in[42];
  float* out = (float*)d_out;

  char* base = (char*)d_ws;
  size_t off = 0;
  auto alloc = [&](size_t bytes) -> void* {
    void* p = base + off;
    off += (bytes + 255) & ~(size_t)255;
    return p;
  };
  _Float16* w_c0 = (_Float16*)alloc(128 * 64 * 2);
  _Float16* w_res[6]; for (int i = 0; i < 6; ++i) w_res[i] = (_Float16*)alloc(128 * 1152 * 2);
  _Float16* w_c1   = (_Float16*)alloc(16 * 128 * 2);
  _Float16* w_rwih = (_Float16*)alloc(768 * 800 * 2);
  _Float16* w_rwhh = (_Float16*)alloc(768 * 256 * 2);
  _Float16* w_q    = (_Float16*)alloc(128 * 256 * 2);
  _Float16* w_k    = (_Float16*)alloc(128 * 256 * 2);
  _Float16* w_v    = (_Float16*)alloc(128 * 256 * 2);
  _Float16* w_o    = (_Float16*)alloc(64 * 128 * 2);
  _Float16* w_uwih = (_Float16*)alloc(768 * 64 * 2);
  _Float16* w_uwhh = (_Float16*)alloc(768 * 256 * 2);
  _Float16* actA   = (_Float16*)alloc((size_t)NS * CNN * 49 * 2);
  _Float16* actB   = (_Float16*)alloc((size_t)NS * CNN * 49 * 2);
  _Float16* lat16  = (_Float16*)alloc((size_t)NS * LATP * 2);
  float* gi  = (float*)alloc((size_t)ROWS * 768 * 4);
  float* gh  = (float*)alloc((size_t)ROWS * 768 * 4);
  float* qb  = (float*)alloc((size_t)ROWS * 128 * 4);
  float* kb  = (float*)alloc((size_t)ROWS * 128 * 4);
  float* vb  = (float*)alloc((size_t)ROWS * 128 * 4);
  float* ctx = (float*)alloc((size_t)ROWS * 128 * 4);
  _Float16* ctx16 = (_Float16*)alloc((size_t)ROWS * 128 * 2);
  float* info = (float*)alloc((size_t)ROWS * 64 * 4);
  _Float16* info16 = (_Float16*)alloc((size_t)ROWS * 64 * 2);
  float* h = (float*)alloc((size_t)ROWS * HID * 4);
  _Float16* h16 = (_Float16*)alloc((size_t)ROWS * HID * 2);
  float* hbuf = (float*)alloc((size_t)SS * BB * HID * 4);

  auto cvt = [&](const float* src, _Float16* dst, int rows, int K, int ldk) {
    int total = rows * ldk;
    k_cvt<<<(total + 255) / 256, 256, 0, stream>>>(src, dst, rows, K, ldk);
  };
  cvt(conv0_w, w_c0, 128, 54, 64);
  for (int i = 0; i < 6; ++i) cvt(res_w[i], w_res[i], 128, 1152, 1152);
  cvt(conv1_w, w_c1, 16, 128, 128);
  cvt(rec_wih, w_rwih, 768, 784, 800);
  cvt(rec_whh, w_rwhh, 768, 256, 256);
  cvt(wq_w, w_q, 128, 256, 256);
  cvt(wk_w, w_k, 128, 256, 256);
  cvt(wv_w, w_v, 128, 256, 256);
  cvt(wo_w, w_o, 64, 128, 128);
  cvt(upd_wih, w_uwih, 768, 64, 64);
  cvt(upd_whh, w_uwhh, 768, 256, 256);
  cvt(conv0_w /*never read (K=0)*/, lat16, NS, 0, LATP);   // zero latent incl. pad

  // ---- conv encoder ----
  k_conv0<<<NCOLS / 64, 256, 0, stream>>>(obs, w_c0, conv0_b, actA);
  for (int r = 0; r < 3; ++r) {
    k_conv3x3<<<NCOLS / 64, 256, 0, stream>>>(w_res[2 * r],     res_b[2 * r],     actA, nullptr, actB);
    k_conv3x3<<<NCOLS / 64, 256, 0, stream>>>(w_res[2 * r + 1], res_b[2 * r + 1], actB, actA,    actA);
  }
  k_conv1x1<<<NCOLS / (64 * 8), 256, 0, stream>>>(w_c1, conv1_b, actA, lat16);

  // ---- recurrence ----
  k_inith<<<(ROWS * HID + 255) / 256, 256, 0, stream>>>(hidden, h, h16);

  auto gemm = [&](const _Float16* A, int lda, const _Float16* B, int ldb,
                  const float* bias, float* C, int ldc, _Float16* C16, int ldc16,
                  int M, int N, int K, int relu) {
    int tm = M / 16, tn = N / 64, total = tm * tn;
    k_gemm<<<(total + 7) / 8, 256, 0, stream>>>(A, lda, B, ldb, bias, C, ldc, C16, ldc16, tm, tn, K, relu);
  };

  for (int s = 0; s < SS; ++s) {
    const unsigned char* cmS = cm + s * 16;
    // recurrent GRU: gi = W_ih * lat_s, gh = W_hh * h
    gemm(w_rwih, 800, lat16 + s * 800, 16 * 800, rec_bih, gi, 768, nullptr, 0, 768, ROWS, 800, 0);
    gemm(w_rwhh, 256, h16, 256, rec_bhh, gh, 768, nullptr, 0, 768, ROWS, 256, 0);
    k_gru<<<(ROWS * HID + 255) / 256, 256, 0, stream>>>(gi, gh, h, h16, cmS, 0);
    for (int l = 0; l < 2; ++l) {
      gemm(w_q, 256, h16, 256, wq_b, qb, 128, nullptr, 0, 128, ROWS, 256, 0);
      gemm(w_k, 256, h16, 256, wk_b, kb, 128, nullptr, 0, 128, ROWS, 256, 0);
      gemm(w_v, 256, h16, 256, wv_b, vb, 128, nullptr, 0, 128, ROWS, 256, 0);
      k_attn<<<BB * 2, 256, 0, stream>>>(qb, kb, vb, cmS, ctx, ctx16);
      gemm(w_o, 128, ctx16, 128, nullptr, info, 64, info16, 64, 64, ROWS, 128, 0);
      gemm(w_uwih, 64, info16, 64, upd_bih, gi, 768, nullptr, 0, 768, ROWS, 64, 0);
      gemm(w_uwhh, 256, h16, 256, upd_bhh, gh, 768, nullptr, 0, 768, ROWS, 256, 0);
      k_gru<<<(ROWS * HID + 255) / 256, 256, 0, stream>>>(gi, gh, h, h16, cmS, 1);
    }
    k_hbuf<<<(BB * HID + 255) / 256, 256, 0, stream>>>(h, hbuf, s);
  }

  // ---- heads ----
  k_head<<<BB, HID, 0, stream>>>(lat16, hbuf, steps, mlp0_w, mlp0_b, mlp1_w, mlp1_b,
                                 adv_w, adv_b, state_w, state_b, out);
}
